// NIF_60224031424766
// MI455X (gfx1250) — matrix operand order repaired and checked by reading, not compiled, not run on hardware
//
#include <hip/hip_runtime.h>

// ---------- CDNA5 WMMA types ----------
typedef __attribute__((ext_vector_type(16))) __bf16   v16bf;
typedef __attribute__((ext_vector_type(8)))  float    v8f;
typedef __attribute__((ext_vector_type(4)))  float    v4f;
typedef __attribute__((ext_vector_type(4)))  unsigned uint4v;

union ABu { v16bf v; unsigned u[8]; uint4v q[2]; };

__device__ __forceinline__ v8f wmma_bf16(v16bf a, v16bf b, v8f c) {
    // D = A(16x32 bf16) * B(32x16 bf16) + C(16x16 f32)
    return __builtin_amdgcn_wmma_f32_16x16x32_bf16(false, a, false, b,
                                                   (short)0, c, false, false);
}

__device__ __forceinline__ float swishf(float x) {
    return x / (1.0f + __expf(-x));
}

// LDS byte-offset of a generic pointer derived from __shared__ memory:
// generic LDS addresses carry the offset in addr[31:0] (flat->LDS truncation).
__device__ __forceinline__ unsigned lds_off(const void* p) {
    return (unsigned)(unsigned long long)p;
}

// Async global->LDS copy (ASYNCcnt-tracked). Each lane moves 16B per issue.
__device__ __forceinline__ void async_copy(unsigned ldsDst, const void* gsrc,
                                           int bytes, int tid) {
    const char* s = (const char*)gsrc;
    for (int o = tid * 16; o < bytes; o += 256 * 16) {
        unsigned l = ldsDst + (unsigned)o;
        asm volatile("global_load_async_to_lds_b128 %0, %1, %2 offset:0"
                     :: "v"(l), "v"(o), "s"(s) : "memory");
    }
}
__device__ __forceinline__ void wait_async0() {
    asm volatile("s_wait_asynccnt 0x0" ::: "memory");
}

// ---------- problem constants ----------
#define B_ROWS   8192
#define TILE_R   128
#define NWG      (B_ROWS / TILE_R)   // 64
#define NST      256
#define NHID     32                  // == WMMA K for bf16, exact fit
#define NSX      64
#define LSX      4
#define LST      4
#define PO_DIM   16897
#define PO_WI    128
#define PO_WL    16512
#define PO_B1    16576
#define PO_BI    16640
#define PO_BL    16896

// ---------- workspace layout (bytes) ----------
#define WS_LAT32 0            // 8192*32*4          = 1048576
#define WS_WH    1048576      // 4 * 65536 bf16     =  524288
#define WS_WB    1572864      // 8192 bf16          =   16384
#define WS_WL    1589248      // 4 * 131072 bf16    = 1048576
#define WS_BT    2637824      // 4 * 2048 bf16      =   16384
#define WS_BLM   2654208      // 4 * 4096 bf16      =   32768

// ======================================================================
// Kernel 0: one-time conversion of all weights to bf16 in the pre-swizzled
// B-operand LDS image layout (lane = {K-half, N}; 16 contiguous halfwords).
// ======================================================================
__global__ __launch_bounds__(256) void nif_prep(
    const float* __restrict__ Wh, const float* __restrict__ Wb,
    const float* __restrict__ Wl, const float* __restrict__ bl,
    char* __restrict__ ws)
{
    __bf16* whbf = (__bf16*)(ws + WS_WH);
    __bf16* wbbf = (__bf16*)(ws + WS_WB);
    __bf16* wlbf = (__bf16*)(ws + WS_WL);
    __bf16* btbf = (__bf16*)(ws + WS_BT);
    __bf16* blbf = (__bf16*)(ws + WS_BLM);
    const int g = blockIdx.x * blockDim.x + threadIdx.x;
    const int stride = gridDim.x * blockDim.x;

    // Wh: 4 layers of 256x256, K-blocked (kb) tiles of 32x16
    for (int e = g; e < 4 * 65536; e += stride) {
        int l = e >> 16, r = e & 65535, k = r >> 8, c = r & 255;
        whbf[l * 65536 + ((k >> 5) * 16 + (c >> 4)) * 512 +
             ((k >> 4) & 1) * 256 + (c & 15) * 16 + (k & 15)] = (__bf16)Wh[e];
    }
    // Wb: 256x32
    for (int e = g; e < 8192; e += stride) {
        int k = e >> 5, c = e & 31;
        wbbf[((k >> 5) * 2 + (c >> 4)) * 512 + ((k >> 4) & 1) * 256 +
             (c & 15) * 16 + (k & 15)] = (__bf16)Wb[e];
    }
    // Wl field-weight blocks: 4 layers of 32x4096 (K=32 -> single k-step)
    for (int e = g; e < 4 * 131072; e += stride) {
        int l = e >> 17, r = e & 131071, k = r >> 12, c = r & 4095;
        wlbf[l * 131072 + (c >> 4) * 512 + (k >> 4) * 256 +
             (c & 15) * 16 + (k & 15)] = (__bf16)Wl[k * PO_DIM + PO_WI + l * 4096 + c];
    }
    // b_i generator columns: 4 layers of 32x64
    for (int e = g; e < 4 * 2048; e += stride) {
        int l = e >> 11, r = e & 2047, k = r >> 6, c = r & 63;
        btbf[l * 2048 + (c >> 4) * 512 + (k >> 4) * 256 +
             (c & 15) * 16 + (k & 15)] = (__bf16)Wl[k * PO_DIM + PO_BI + l * 64 + c];
    }
    // BL matrices (bl of w_i blocks reshaped 64x64), K=64 -> two k-steps
    for (int e = g; e < 4 * 4096; e += stride) {
        int l = e >> 12, r = e & 4095, i = r >> 6, j = r & 63;
        blbf[l * 4096 + ((i >> 5) * 4 + (j >> 4)) * 512 + ((i >> 4) & 1) * 256 +
             (j & 15) * 16 + (i & 15)] = (__bf16)bl[PO_WI + l * 4096 + i * 64 + j];
    }
}

// ======================================================================
// Kernel A: latent network -> latent32 (B x 32, f32) in workspace
// ======================================================================
#define KA_LATA   0
#define KA_LATB   (TILE_R * NST * 2)           // 65536
#define KA_WST    (2 * TILE_R * NST * 2)       // 131072
#define KA_BIAS   (KA_WST + NST * NST * 2)     // 262144
#define KA_LDS    (KA_BIAS + NST * 4)          // 263168 bytes

__global__ __launch_bounds__(256) void nif_latent(
    const float* __restrict__ inputs, const float* __restrict__ W1,
    const float* __restrict__ b1,     const float* __restrict__ bh,
    const float* __restrict__ bb,     char* __restrict__ ws)
{
    extern __shared__ char smem[];
    __bf16* latA  = (__bf16*)(smem + KA_LATA);
    __bf16* latB  = (__bf16*)(smem + KA_LATB);
    __bf16* wst   = (__bf16*)(smem + KA_WST);
    float*  biasf = (float*) (smem + KA_BIAS);

    const int tid   = threadIdx.x;
    const int lane  = tid & 31;
    const int wave  = tid >> 5;
    const int R0    = blockIdx.x * TILE_R;
    const int khalf = lane >> 4;
    const int n15   = lane & 15;
    const int r0    = wave * 16;
    const int rr    = r0 + n15;
    const unsigned wstOff  = lds_off(wst);
    const unsigned biasOff = lds_off(biasf);
    float* lat32g = (float*)(ws + WS_LAT32);

    // latent0 = swish(p * W1 + b1)
    for (int e = tid; e < TILE_R * NST; e += 256) {
        int r = e >> 8, c = e & 255;
        float x = inputs[(R0 + r) * 3 + 0] * W1[c] + b1[c];
        latA[r * NST + c] = (__bf16)swishf(x);
    }
    __syncthreads();

    for (int layer = 0; layer < LST; ++layer) {
        // async-stage this layer's pre-swizzled weights + f32 bias
        async_copy(wstOff,  ws + WS_WH + layer * 131072, 131072, tid);
        async_copy(biasOff, bh + layer * NST, NST * 4, tid);
        wait_async0();
        __syncthreads();

        __bf16* src = (layer & 1) ? latB : latA;
        __bf16* dst = (layer & 1) ? latA : latB;

        ABu a[8];                       // wave's 16 rows x K=256 A tiles
        #pragma unroll
        for (int kb = 0; kb < 8; ++kb) {
            const unsigned* p0 = (const unsigned*)(src + rr * NST + kb * 32 + khalf * 16);
            const unsigned* p1 = (const unsigned*)(src + rr * NST + kb * 32 + 8 + khalf * 16);
            #pragma unroll
            for (int t = 0; t < 4; ++t) { a[kb].u[t] = p0[t]; a[kb].u[4 + t] = p1[t]; }
        }

        for (int ct = 0; ct < 16; ++ct) {
            v8f acc = {};
            #pragma unroll
            for (int kb = 0; kb < 8; ++kb) {
                ABu b;
                const uint4v* bp = (const uint4v*)((char*)wst +
                    (kb * 16 + ct) * 1024 + khalf * 512 + n15 * 32);
                b.q[0] = bp[0]; b.q[1] = bp[1];
                acc = wmma_bf16(a[kb].v, b.v, acc);
            }
            int col = ct * 16 + n15;
            float bias = biasf[col];
            #pragma unroll
            for (int p = 0; p < 8; ++p) {
                int row = r0 + p + 8 * khalf;
                float res = (float)src[row * NST + col];
                dst[row * NST + col] = (__bf16)(res + swishf(acc[p] + bias));
            }
        }
        __syncthreads();
    }

    // latent32 = latent @ Wb + bb   (final latent lives in latA)
    async_copy(wstOff,  ws + WS_WB, 16384, tid);
    async_copy(biasOff, bb, NHID * 4, tid);
    wait_async0();
    __syncthreads();

    for (int ct = 0; ct < 2; ++ct) {
        v8f acc = {};
        #pragma unroll
        for (int kb = 0; kb < 8; ++kb) {
            ABu a, b;
            const unsigned* p0 = (const unsigned*)(latA + rr * NST + kb * 32 + khalf * 16);
            const unsigned* p1 = (const unsigned*)(latA + rr * NST + kb * 32 + 8 + khalf * 16);
            #pragma unroll
            for (int t = 0; t < 4; ++t) { a.u[t] = p0[t]; a.u[4 + t] = p1[t]; }
            const uint4v* bp = (const uint4v*)((char*)wst +
                (kb * 2 + ct) * 1024 + khalf * 512 + n15 * 32);
            b.q[0] = bp[0]; b.q[1] = bp[1];
            acc = wmma_bf16(a.v, b.v, acc);
        }
        int col = ct * 16 + n15;
        float bias = biasf[col];
        #pragma unroll
        for (int p = 0; p < 8; ++p)
            lat32g[(R0 + r0 + p + 8 * khalf) * NHID + col] = acc[p] + bias;
    }
}

// ======================================================================
// Kernel B: fused hypernet weight generation + field net
//   - weight tiles generated by WMMA and contracted on the fly (FMA only;
//     bl contribution folded into a u @ BL GEMM done with WMMA)
//   - double-buffered 64KB quarters staged with async global->LDS
// ======================================================================
#define KB_BUF0   0
#define KB_BUF1   65536
#define KB_BIAST  131072                        // 4096 B
#define KB_BLST   (KB_BIAST + 4096)             // 135168, 8192 B
#define KB_U      (KB_BLST + 8192)              // 143360, 32768 B
#define KB_L32    (KB_U + TILE_R * NSX * 4)     // 176128, 16384 B
#define KB_LDS    (KB_L32 + TILE_R * NHID * 4)  // 192512 bytes

__global__ __launch_bounds__(256) void nif_field(
    const float* __restrict__ inputs, const float* __restrict__ Wl,
    const float* __restrict__ bl,     char* __restrict__ ws,
    float* __restrict__ out)
{
    extern __shared__ char smem[];
    char*   buf0  = smem + KB_BUF0;              // NOTE: scalar pointers, no
    char*   buf1  = smem + KB_BUF1;              // pointer-array initializer
    float*  wldsf = (float*)(smem + KB_BUF0);    // f32 alias for head/tail staging
    char*   biasT = smem + KB_BIAST;
    char*   BLst  = smem + KB_BLST;
    float*  ubuf  = (float*)(smem + KB_U);
    float*  l32   = (float*)(smem + KB_L32);

    const int tid   = threadIdx.x;
    const int lane  = tid & 31;
    const int wave  = tid >> 5;
    const int R0    = blockIdx.x * TILE_R;
    const int khalf = lane >> 4;
    const int n15   = lane & 15;
    const int r0    = wave * 16;
    const int rr    = r0 + n15;
    const float* lat32g = (const float*)(ws + WS_LAT32);

    // ---- stage latent32 tile + head generator columns (f32) ----
    for (int e = tid; e < TILE_R * NHID; e += 256) l32[e] = lat32g[R0 * NHID + e];
    for (int e = tid; e < 32 * 192; e += 256) {
        int k = e / 192, c = e % 192;
        int gc = (c < 128) ? c : (PO_B1 + c - 128);
        wldsf[e] = Wl[k * PO_DIM + gc];
    }
    __syncthreads();

    // ---- head: u = swish(s @ w1 + b1) generated per row (VALU) ----
    for (int e = tid; e < TILE_R * NSX; e += 256) {
        int r = e >> 6, j = e & 63;
        const float* lr = l32 + r * NHID;
        float a0 = 0.f, a1 = 0.f, ab = 0.f;
        #pragma unroll 8
        for (int k = 0; k < NHID; ++k) {
            float lk = lr[k];
            a0 += lk * wldsf[k * 192 + j];
            a1 += lk * wldsf[k * 192 + 64 + j];
            ab += lk * wldsf[k * 192 + 128 + j];
        }
        a0 += bl[j]; a1 += bl[64 + j]; ab += bl[PO_B1 + j];
        float s0 = inputs[(R0 + r) * 3 + 1];
        float s1 = inputs[(R0 + r) * 3 + 2];
        ubuf[e] = swishf(s0 * a0 + s1 * a1 + ab);
    }

    // A operand: wave's 16 latent32 rows, bf16, K=32 (resident all kernel)
    ABu alat;
    {
        const float* lr = l32 + rr * NHID;
        #pragma unroll
        for (int e16 = 0; e16 < 16; ++e16) {
            int k = khalf * 16 + e16;
            alat.v[e16] = (__bf16)lr[k];
        }
    }
    __syncthreads();

    for (int layer = 0; layer < LSX; ++layer) {
        // stage quarter 0 + b_i generators + BL matrix (async)
        async_copy(lds_off(buf0),  ws + WS_WL + layer * 262144, 65536, tid);
        async_copy(lds_off(biasT), ws + WS_BT + layer * 4096, 4096, tid);
        async_copy(lds_off(BLst),  ws + WS_BLM + layer * 8192, 8192, tid);
        wait_async0();
        __syncthreads();

        // u as bf16 A operands (K=64 -> two k-steps); wave-local rows
        ABu au0, au1;
        {
            const float* ur = ubuf + rr * NSX;
            #pragma unroll
            for (int e16 = 0; e16 < 16; ++e16) {
                int k = khalf * 16 + e16;
                au0.v[e16] = (__bf16)ur[k];
                au1.v[e16] = (__bf16)ur[32 + k];
            }
        }

        // z init = generated b_i  +  u @ BL (bl folded via WMMA)  + bl(b_i)
        v8f zacc[4];
        #pragma unroll
        for (int jc = 0; jc < 4; ++jc) {
            ABu b;
            const uint4v* bp = (const uint4v*)(biasT + jc * 1024 + khalf * 512 + n15 * 32);
            b.q[0] = bp[0]; b.q[1] = bp[1];
            v8f z = {};
            z = wmma_bf16(alat.v, b.v, z);
            const uint4v* l0 = (const uint4v*)(BLst + (0 * 4 + jc) * 1024 + khalf * 512 + n15 * 32);
            ABu b0; b0.q[0] = l0[0]; b0.q[1] = l0[1];
            z = wmma_bf16(au0.v, b0.v, z);
            const uint4v* l1 = (const uint4v*)(BLst + (1 * 4 + jc) * 1024 + khalf * 512 + n15 * 32);
            ABu b1v; b1v.q[0] = l1[0]; b1v.q[1] = l1[1];
            z = wmma_bf16(au1.v, b1v.v, z);
            float bv = bl[PO_BI + layer * 64 + jc * 16 + n15];
            #pragma unroll
            for (int p = 0; p < 8; ++p) zacc[jc][p] = z[p] + bv;
        }

        // quarters: async-prefetch next 64KB while WMMA-ing current one
        for (int q = 0; q < 4; ++q) {
            if (q < 3)
                async_copy(lds_off(((q + 1) & 1) ? buf1 : buf0),
                           ws + WS_WL + layer * 262144 + (q + 1) * 65536, 65536, tid);
            const char* bc = (q & 1) ? buf1 : buf0;
            for (int lg = 0; lg < 4; ++lg) {         // 4 i-indices per group
                v4f us4[8];
                #pragma unroll
                for (int p = 0; p < 8; ++p)
                    us4[p] = *(const v4f*)&ubuf[(r0 + p + 8 * khalf) * NSX + q * 16 + lg * 4];
                #pragma unroll
                for (int t = 0; t < 4; ++t) {
                    int li = lg * 4 + t;
                    #pragma unroll
                    for (int jc = 0; jc < 4; ++jc) {
                        ABu b;
                        const uint4v* bp = (const uint4v*)(bc +
                            (li * 4 + jc) * 1024 + khalf * 512 + n15 * 32);
                        b.q[0] = bp[0]; b.q[1] = bp[1];
                        v8f d = {};
                        d = wmma_bf16(alat.v, b.v, d);   // W_gen 16x16 tile
                        #pragma unroll
                        for (int p = 0; p < 8; ++p)
                            zacc[jc][p] += us4[p][t] * d[p];
                    }
                }
            }
            wait_async0();
            __syncthreads();
        }

        // u = swish(z) + u   (wave-local rows, no barrier needed)
        #pragma unroll
        for (int jc = 0; jc < 4; ++jc) {
            #pragma unroll
            for (int p = 0; p < 8; ++p) {
                int row = r0 + p + 8 * khalf;
                int j   = jc * 16 + n15;
                ubuf[row * NSX + j] = swishf(zacc[jc][p]) + ubuf[row * NSX + j];
            }
        }
    }

    // ---- tail: out = u @ w_l + b_l (generated per row, VALU) ----
    __syncthreads();
    for (int e = tid; e < 32 * 65; e += 256) {
        int k = e / 65, c = e % 65;
        int gc = (c < 64) ? (PO_WL + c) : PO_BL;
        wldsf[e] = Wl[k * PO_DIM + gc];
    }
    __syncthreads();
    if (tid < TILE_R) {
        int r = tid;
        const float* lr = l32 + r * NHID;
        float acc = 0.f;
        for (int j = 0; j < NSX; ++j) {
            float w = 0.f;
            #pragma unroll 8
            for (int k = 0; k < NHID; ++k) w += lr[k] * wldsf[k * 65 + j];
            acc += ubuf[r * NSX + j] * (w + bl[PO_WL + j]);
        }
        float blv = 0.f;
        #pragma unroll 8
        for (int k = 0; k < NHID; ++k) blv += lr[k] * wldsf[k * 65 + 64];
        out[R0 + r] = acc + blv + bl[PO_BL];
    }
}

// ======================================================================
extern "C" void kernel_launch(void* const* d_in, const int* in_sizes, int n_in,
                              void* d_out, int out_size, void* d_ws, size_t ws_size,
                              hipStream_t stream)
{
    (void)in_sizes; (void)n_in; (void)out_size; (void)ws_size;
    const float* inputs = (const float*)d_in[0];
    const float* W1     = (const float*)d_in[1];
    const float* b1     = (const float*)d_in[2];
    const float* Wh     = (const float*)d_in[3];
    const float* bh     = (const float*)d_in[4];
    const float* Wb     = (const float*)d_in[5];
    const float* bb     = (const float*)d_in[6];
    const float* Wl     = (const float*)d_in[7];
    const float* bl     = (const float*)d_in[8];
    char*  ws   = (char*)d_ws;
    float* outp = (float*)d_out;

    nif_prep  <<<512, 256, 0, stream>>>(Wh, Wb, Wl, bl, ws);
    nif_latent<<<NWG, 256, KA_LDS, stream>>>(inputs, W1, b1, bh, bb, ws);
    nif_field <<<NWG, 256, KB_LDS, stream>>>(inputs, Wl, bl, ws, outp);
}
